// GTrXLCell_90718299226279
// MI455X (gfx1250) — matrix operand order repaired and checked by reading, not compiled, not run on hardware
//
#include <hip/hip_runtime.h>
#include <hip/hip_bf16.h>
#include <math.h>

typedef __attribute__((ext_vector_type(16))) __bf16 bf16x16;
typedef __attribute__((ext_vector_type(8)))  float  f32x8;
typedef __attribute__((ext_vector_type(4)))  unsigned uint32x4;
typedef __attribute__((ext_vector_type(8)))  int      int32x8;
typedef __attribute__((ext_vector_type(4)))  int      int32x4;

#define B_SZ   4096
#define D_SZ   1024
#define S_SZ   17
#define H_SZ   8
#define HD_SZ  128
#define DIN_SZ 512

#define USE_TDM 1

// ---------------------------------------------------------------- convert f32 -> bf16
__global__ void cvt_bf16_kernel(const float* __restrict__ src, __bf16* __restrict__ dst, int n) {
    int i = blockIdx.x * blockDim.x + threadIdx.x;
    int stride = gridDim.x * blockDim.x;
    for (; i < n; i += stride) dst[i] = (__bf16)src[i];
}

// ---------------------------------------------------------------- LayerNorm (+optional mem_next copy)
__global__ void ln_kernel(const float* __restrict__ src_mem, const float* __restrict__ src_x,
                          const float* __restrict__ g, const float* __restrict__ bta,
                          __bf16* __restrict__ dst, float* __restrict__ copy_dst, int mode) {
    const int row = blockIdx.x;
    const int t = threadIdx.x;
    const float* src;
    float* cp = nullptr;
    if (mode == 0) {
        int s = row >> 12, b = row & 4095;
        src = (s < 16) ? (src_mem + (size_t)row * D_SZ) : (src_x + (size_t)b * D_SZ);
        if (s >= 1) cp = copy_dst + (size_t)(row - B_SZ) * D_SZ;
    } else {
        src = src_mem + (size_t)row * D_SZ;
    }
    __shared__ float red[256];
    float4 xv = *(const float4*)(src + t * 4);
    if (cp) *(float4*)(cp + t * 4) = xv;

    red[t] = xv.x + xv.y + xv.z + xv.w;
    __syncthreads();
    for (int st = 128; st > 0; st >>= 1) { if (t < st) red[t] += red[t + st]; __syncthreads(); }
    float mean = red[0] * (1.0f / D_SZ);
    __syncthreads();

    float d0 = xv.x - mean, d1 = xv.y - mean, d2 = xv.z - mean, d3 = xv.w - mean;
    red[t] = d0 * d0 + d1 * d1 + d2 * d2 + d3 * d3;
    __syncthreads();
    for (int st = 128; st > 0; st >>= 1) { if (t < st) red[t] += red[t + st]; __syncthreads(); }
    float inv = rsqrtf(red[0] * (1.0f / D_SZ) + 1e-5f);

    const float* xp = &xv.x;
#pragma unroll
    for (int i = 0; i < 4; i++) {
        int c = t * 4 + i;
        float v = (xp[i] - mean) * inv * g[c] + bta[c];
        dst[(size_t)row * D_SZ + c] = (__bf16)v;
    }
}

// ---------------------------------------------------------------- TDM: DMA a 128x32 bf16 tile into LDS
// D# per CDNA5 ISA ch.8: 2D tile, data_size=2B, row stride = K elements.
// LDS padding: pad_interval=3 (every 16 dwords = one 32-bf16 row), pad_amount=3 (4 dwords = 16B)
// -> LDS row stride of 40 bf16, matching the compute-side padded layout.
__device__ __forceinline__ void tdm_load_tile(const void* gptr, unsigned lds_off, int K) {
    unsigned long long ga = (unsigned long long)(uintptr_t)gptr;
    uint32x4 g0 = { 1u,                                    // count=1 (valid), load, user mode
                    lds_off,                               // lds_addr (bytes)
                    (unsigned)ga,                          // global_addr[31:0]
                    (unsigned)((ga >> 32) & 0x01FFFFFFull) // global_addr[56:32]
                        | 0x80000000u };                   // type=2 ("image")
    int32x8 g1 = { (int)0x06D10000u,                       // data_size=2B | pad_en | intv=16dw | amt=4dw
                   (int)((unsigned)(K & 0xFFFF) << 16),    // tensor_dim0[15:0] (atomic_addr=0)
                   (int)((K >> 16) & 0xFFFF),              // tensor_dim0[31:16] | tensor_dim1[15:0]=0
                   (int)(16u | (32u << 16)),               // tensor_dim1[31:16]=16 (dim1=1M) | tile_dim0=32
                   (int)128,                               // tile_dim1=128 | tile_dim2=0
                   (int)K,                                 // tensor_dim0_stride[31:0] = K elements
                   0, 0 };
    int32x4 gz = { 0, 0, 0, 0 };
#if defined(__clang_major__) && __clang_major__ >= 23
    int32x8 gz8 = { 0, 0, 0, 0, 0, 0, 0, 0 };
    __builtin_amdgcn_tensor_load_to_lds(g0, g1, gz, gz, gz8, 0);
#else
    __builtin_amdgcn_tensor_load_to_lds(g0, g1, gz, gz, 0);
#endif
}

// ---------------------------------------------------------------- BF16 WMMA GEMM: C = A @ W^T + bias (+epilogue)
// EPI: 0=f32 store | 2=qkv split store | 3=gate->f32 | 4=gelu->bf16 | 5=residual->f32
template <int EPI>
__global__ void gemm_bf16_kernel(const __bf16* __restrict__ A, const __bf16* __restrict__ Wt,
                                 const float* __restrict__ bias,
                                 float* __restrict__ outF, __bf16* __restrict__ outB,
                                 __bf16* __restrict__ q_ws, __bf16* __restrict__ k_ws, __bf16* __restrict__ v_ws,
                                 const float* __restrict__ aux0, const float* __restrict__ aux1,
                                 const float* __restrict__ aux2,
                                 int M, int N, int K) {
    constexpr int BM = 128, BN = 128, BK = 32, LDSS = BK + 8; // padded stride (bf16 units)
    const int mBase = blockIdx.y * BM;
    const int nBase = blockIdx.x * BN;

    if (EPI == 2) { // skip pure-Q column tiles for s < 16 (only last position's q is used)
        if (nBase + BN <= D_SZ && (mBase >> 12) != 16) return;
    }

    __shared__ __bf16 sA[2][BM * LDSS];
    __shared__ __bf16 sB[2][BN * LDSS];

    const int tid  = threadIdx.x;
    const int wave = tid >> 5, lane = tid & 31;
    const int wm = wave & 3;   // wave row:   wm*32
    const int wn = wave >> 2;  // wave col:   wn*64
    const int lm = lane & 15, lhi = lane >> 4;

    f32x8 acc[2][4] = {};

    const int nIter = K / BK;

#if USE_TDM
    const unsigned ldsA0 = (unsigned)(uintptr_t)&sA[0][0];
    const unsigned ldsA1 = (unsigned)(uintptr_t)&sA[1][0];
    const unsigned ldsB0 = (unsigned)(uintptr_t)&sB[0][0];
    const unsigned ldsB1 = (unsigned)(uintptr_t)&sB[1][0];
    if (wave == 0) { // one wave drives the Tensor Data Mover
        tdm_load_tile(A  + (size_t)mBase * K, ldsA0, K);
        tdm_load_tile(Wt + (size_t)nBase * K, ldsB0, K);
    }
#endif

    for (int it = 0; it < nIter; ++it) {
        const int cur = it & 1;
        const int k0 = it * BK;
#if USE_TDM
        if (wave == 0) {
            if (it + 1 < nIter) { // prefetch next tile while current is consumed
                tdm_load_tile(A  + (size_t)mBase * K + (k0 + BK), cur ? ldsA0 : ldsA1, K);
                tdm_load_tile(Wt + (size_t)nBase * K + (k0 + BK), cur ? ldsB0 : ldsB1, K);
                __builtin_amdgcn_s_wait_tensorcnt(2); // current tile's 2 loads retired (in-order)
            } else {
                __builtin_amdgcn_s_wait_tensorcnt(0);
            }
        }
        __syncthreads();
#else
#pragma unroll
        for (int p = 0; p < 2; p++) {
            int idx = tid + p * 256;
            int row = idx >> 2, c16 = idx & 3;
            *(uint4*)&sA[cur][row * LDSS + c16 * 8] =
                *(const uint4*)(A + (size_t)(mBase + row) * K + k0 + c16 * 8);
            *(uint4*)&sB[cur][row * LDSS + c16 * 8] =
                *(const uint4*)(Wt + (size_t)(nBase + row) * K + k0 + c16 * 8);
        }
        __syncthreads();
#endif
        // B fragments (32x16 KxN): lane: n=lane%16; K = (lane/16)*16 + 2v + h
        bf16x16 bfrag[4];
#pragma unroll
        for (int ni = 0; ni < 4; ni++) {
            union { unsigned u[8]; bf16x16 v; } bu;
            int n = wn * 64 + ni * 16 + lm;
#pragma unroll
            for (int v = 0; v < 8; v++)
                bu.u[v] = *(const unsigned*)&sB[cur][n * LDSS + (lhi * 16 + v * 2)];
            bfrag[ni] = bu.v;
        }
        // A fragments (16x32 MxK): lane: m=lane%16; K = (lane/16)*16 + 2v + h
#pragma unroll
        for (int mi = 0; mi < 2; mi++) {
            union { unsigned u[8]; bf16x16 v; } au;
            int m = wm * 32 + mi * 16 + lm;
#pragma unroll
            for (int v = 0; v < 8; v++) {
                int kk = lhi * 16 + v * 2;
                au.u[v] = *(const unsigned*)&sA[cur][m * LDSS + kk];
            }
#pragma unroll
            for (int ni = 0; ni < 4; ni++)
                acc[mi][ni] = __builtin_amdgcn_wmma_f32_16x16x32_bf16(
                    false, au.v, false, bfrag[ni], (short)0, acc[mi][ni], false, false);
        }
        __syncthreads(); // all waves done reading `cur` before TDM refills it
    }

    // epilogue: C element (m = r + 8*(lane/16), n = lane%16) per 16x16 frag
#pragma unroll
    for (int mi = 0; mi < 2; mi++) {
#pragma unroll
        for (int ni = 0; ni < 4; ni++) {
            f32x8 c = acc[mi][ni];
            int col = nBase + wn * 64 + ni * 16 + lm;
            float bcol = bias[col];
#pragma unroll
            for (int r = 0; r < 8; r++) {
                int row = mBase + wm * 32 + mi * 16 + lhi * 8 + r;
                float val = c[r] + bcol;
                if (EPI == 0) {
                    outF[(size_t)row * N + col] = val;
                } else if (EPI == 2) {
                    int s = row >> 12, b = row & 4095;
                    if (col < D_SZ) {
                        if (s == 16) q_ws[(size_t)b * D_SZ + col] = (__bf16)val;
                    } else if (col < 2 * D_SZ) {
                        k_ws[(size_t)row * D_SZ + (col - D_SZ)] = (__bf16)val;
                    } else {
                        v_ws[(size_t)row * D_SZ + (col - 2 * D_SZ)] = (__bf16)val;
                    }
                } else if (EPI == 3) {
                    float ga = 1.0f / (1.0f + expf(-aux1[col]));
                    float gb = 1.0f / (1.0f + expf(-aux2[col]));
                    outF[(size_t)row * N + col] = ga * aux0[(size_t)row * N + col] + gb * val;
                } else if (EPI == 4) {
                    float gl = 0.5f * val * (1.0f + erff(val * 0.70710678118654752f));
                    outB[(size_t)row * N + col] = (__bf16)gl;
                } else if (EPI == 5) {
                    outF[(size_t)row * N + col] = aux0[(size_t)row * N + col] + val;
                }
            }
        }
    }
}

// ---------------------------------------------------------------- attention (q only for last position)
__global__ void attn_kernel(const __bf16* __restrict__ q_ws, const __bf16* __restrict__ k_ws,
                            const __bf16* __restrict__ v_ws, __bf16* __restrict__ ctx) {
    int idx = blockIdx.x * blockDim.x + threadIdx.x;  // 0 .. B*H-1
    int h = idx & (H_SZ - 1), b = idx >> 3;
    const float scale = 0.08838834764831845f;          // 1/sqrt(128)
    const __bf16* qp = q_ws + (size_t)b * D_SZ + h * HD_SZ;

    float sc[S_SZ];
#pragma unroll 1
    for (int j = 0; j < S_SZ; j++) {
        const __bf16* kp = k_ws + ((size_t)j * B_SZ + b) * D_SZ + h * HD_SZ;
        float acc = 0.0f;
        for (int d = 0; d < HD_SZ; d++) acc += (float)qp[d] * (float)kp[d];
        sc[j] = acc * scale;
    }
    float mx = sc[0];
#pragma unroll
    for (int j = 1; j < S_SZ; j++) mx = fmaxf(mx, sc[j]);
    float se = 0.0f;
#pragma unroll
    for (int j = 0; j < S_SZ; j++) { sc[j] = expf(sc[j] - mx); se += sc[j]; }
    float inv = 1.0f / se;
#pragma unroll
    for (int j = 0; j < S_SZ; j++) sc[j] *= inv;

#pragma unroll 1
    for (int d = 0; d < HD_SZ; d++) {
        float a = 0.0f;
#pragma unroll
        for (int j = 0; j < S_SZ; j++)
            a += sc[j] * (float)v_ws[((size_t)j * B_SZ + b) * D_SZ + h * HD_SZ + d];
        ctx[(size_t)b * D_SZ + h * HD_SZ + d] = (__bf16)a;
    }
}

// ---------------------------------------------------------------- host side
static inline char* ws_bump(char*& p, size_t bytes) {
    char* r = p;
    p += (bytes + 255) & ~(size_t)255;
    return r;
}

extern "C" void kernel_launch(void* const* d_in, const int* in_sizes, int n_in,
                              void* d_out, int out_size, void* d_ws, size_t ws_size,
                              hipStream_t stream) {
    const float* x_t      = (const float*)d_in[0];
    const float* mem_prev = (const float*)d_in[1];
    const float* W_proj   = (const float*)d_in[2];
    const float* b_proj   = (const float*)d_in[3];
    const float* W_in     = (const float*)d_in[4];
    const float* b_in     = (const float*)d_in[5];
    const float* W_out    = (const float*)d_in[6];
    const float* b_out    = (const float*)d_in[7];
    const float* ln1_g    = (const float*)d_in[8];
    const float* ln1_b    = (const float*)d_in[9];
    const float* ln2_g    = (const float*)d_in[10];
    const float* ln2_b    = (const float*)d_in[11];
    const float* gate_a   = (const float*)d_in[12];
    const float* gate_b   = (const float*)d_in[13];
    const float* W1       = (const float*)d_in[14];
    const float* b1       = (const float*)d_in[15];
    const float* W2       = (const float*)d_in[16];
    const float* b2       = (const float*)d_in[17];

    float* out_main = (float*)d_out;                       // (B, D)
    float* mem_next = out_main + (size_t)B_SZ * D_SZ;      // (16, B, D)

    char* p = (char*)d_ws;
    float*  xproj   = (float*) ws_bump(p, (size_t)B_SZ * D_SZ * 4);
    __bf16* ln_bf   = (__bf16*)ws_bump(p, (size_t)S_SZ * B_SZ * D_SZ * 2);
    __bf16* q_bf    = (__bf16*)ws_bump(p, (size_t)B_SZ * D_SZ * 2);
    __bf16* k_bf    = (__bf16*)ws_bump(p, (size_t)S_SZ * B_SZ * D_SZ * 2);
    __bf16* v_bf    = (__bf16*)ws_bump(p, (size_t)S_SZ * B_SZ * D_SZ * 2);
    __bf16* ctx_bf  = (__bf16*)ws_bump(p, (size_t)B_SZ * D_SZ * 2);
    float*  hhat    = (float*) ws_bump(p, (size_t)B_SZ * D_SZ * 4);
    __bf16* hln_bf  = (__bf16*)ws_bump(p, (size_t)B_SZ * D_SZ * 2);
    __bf16* ff1_bf  = (__bf16*)ws_bump(p, (size_t)B_SZ * 4 * D_SZ * 2);
    __bf16* xt_bf   = (__bf16*)ws_bump(p, (size_t)B_SZ * DIN_SZ * 2);
    __bf16* Wp_bf   = (__bf16*)ws_bump(p, (size_t)D_SZ * DIN_SZ * 2);
    __bf16* Win_bf  = (__bf16*)ws_bump(p, (size_t)3 * D_SZ * D_SZ * 2);
    __bf16* Wo_bf   = (__bf16*)ws_bump(p, (size_t)D_SZ * D_SZ * 2);
    __bf16* W1_bf   = (__bf16*)ws_bump(p, (size_t)4 * D_SZ * D_SZ * 2);
    __bf16* W2_bf   = (__bf16*)ws_bump(p, (size_t)4 * D_SZ * D_SZ * 2);

    // 1) one-time f32 -> bf16 conversions
    cvt_bf16_kernel<<<1024, 256, 0, stream>>>(x_t,    xt_bf,  B_SZ * DIN_SZ);
    cvt_bf16_kernel<<<1024, 256, 0, stream>>>(W_proj, Wp_bf,  D_SZ * DIN_SZ);
    cvt_bf16_kernel<<<1024, 256, 0, stream>>>(W_in,   Win_bf, 3 * D_SZ * D_SZ);
    cvt_bf16_kernel<<<1024, 256, 0, stream>>>(W_out,  Wo_bf,  D_SZ * D_SZ);
    cvt_bf16_kernel<<<1024, 256, 0, stream>>>(W1,     W1_bf,  4 * D_SZ * D_SZ);
    cvt_bf16_kernel<<<1024, 256, 0, stream>>>(W2,     W2_bf,  4 * D_SZ * D_SZ);

    // 2) x_proj = x_t @ W_proj^T + b_proj   (f32 out)
    gemm_bf16_kernel<0><<<dim3(D_SZ / 128, B_SZ / 128), 256, 0, stream>>>(
        xt_bf, Wp_bf, b_proj, xproj, nullptr, nullptr, nullptr, nullptr,
        nullptr, nullptr, nullptr, B_SZ, D_SZ, DIN_SZ);

    // 3) LN1 over mem_cat -> bf16; fused mem_next copy
    ln_kernel<<<S_SZ * B_SZ, 256, 0, stream>>>(mem_prev, xproj, ln1_g, ln1_b,
                                               ln_bf, mem_next, 0);

    // 4) QKV projection (skips q columns except for s==16), split-store bf16
    gemm_bf16_kernel<2><<<dim3(3 * D_SZ / 128, S_SZ * B_SZ / 128), 256, 0, stream>>>(
        ln_bf, Win_bf, b_in, nullptr, nullptr, q_bf, k_bf, v_bf,
        nullptr, nullptr, nullptr, S_SZ * B_SZ, 3 * D_SZ, D_SZ);

    // 5) attention for the last position only
    attn_kernel<<<(B_SZ * H_SZ) / 256, 256, 0, stream>>>(q_bf, k_bf, v_bf, ctx_bf);

    // 6) out-projection + sigmoid-gate epilogue -> h_hat (f32)
    const float* mem_last = mem_prev + (size_t)15 * B_SZ * D_SZ;
    gemm_bf16_kernel<3><<<dim3(D_SZ / 128, B_SZ / 128), 256, 0, stream>>>(
        ctx_bf, Wo_bf, b_out, hhat, nullptr, nullptr, nullptr, nullptr,
        mem_last, gate_a, gate_b, B_SZ, D_SZ, D_SZ);

    // 7) LN2 -> bf16
    ln_kernel<<<B_SZ, 256, 0, stream>>>(hhat, nullptr, ln2_g, ln2_b, hln_bf, nullptr, 1);

    // 8) FF1 + exact GELU -> bf16
    gemm_bf16_kernel<4><<<dim3(4 * D_SZ / 128, B_SZ / 128), 256, 0, stream>>>(
        hln_bf, W1_bf, b1, nullptr, ff1_bf, nullptr, nullptr, nullptr,
        nullptr, nullptr, nullptr, B_SZ, 4 * D_SZ, D_SZ);

    // 9) FF2 + residual add -> d_out (f32)
    gemm_bf16_kernel<5><<<dim3(D_SZ / 128, B_SZ / 128), 256, 0, stream>>>(
        ff1_bf, W2_bf, b2, out_main, nullptr, nullptr, nullptr, nullptr,
        hhat, nullptr, nullptr, B_SZ, D_SZ, 4 * D_SZ);
}
